// SRED_rho_44805098832319
// MI455X (gfx1250) — matrix operand order repaired and checked by reading, not compiled, not run on hardware
//
#include <hip/hip_runtime.h>
#include <math.h>

// ---- problem constants (from reference) ----
#define B_     4
#define LS_    512
#define LW_    512
#define LY_    512
#define M_     16
#define NSTEP_ 10
#define HID_   128
#define D_     2560          // 2*LS + 2*LW + LY
#define ROWS_  (B_ * M_)     // 64 MLP rows

typedef __attribute__((ext_vector_type(16))) _Float16 v16h;
typedef __attribute__((ext_vector_type(8)))  _Float16 v8h;
typedef __attribute__((ext_vector_type(8)))  float    v8f;

// ---------------------------------------------------------------------------
// W1 fp32 -> fp16 (once per launch)
// ---------------------------------------------------------------------------
__global__ void k_w1half(const float* __restrict__ W1, _Float16* __restrict__ w1h, int n) {
    int idx = blockIdx.x * blockDim.x + threadIdx.x;
    if (idx < n) w1h[idx] = (_Float16)W1[idx];
}

// ---------------------------------------------------------------------------
// constant part of MLP input x: cols [1024,2560) = wr, wi, y  (once per launch)
// ---------------------------------------------------------------------------
__global__ void k_xconst(const float* __restrict__ w, const float* __restrict__ y,
                         _Float16* __restrict__ xh) {
    int idx = blockIdx.x * blockDim.x + threadIdx.x;   // 64 * 1536
    if (idx >= ROWS_ * 1536) return;
    int r = idx / 1536, c = idx % 1536;
    int b = r >> 4, m = r & 15;
    float v;
    if (c < 512)       v = w[((size_t)(b * LW_ + c)         * M_ + m) * 2 + 0]; // wr
    else if (c < 1024) v = w[((size_t)(b * LW_ + (c - 512)) * M_ + m) * 2 + 1]; // wi
    else               v = y[(size_t)(c - 1024) * M_ + m];                      // y^T
    xh[(size_t)r * D_ + 1024 + c] = (_Float16)v;
}

__global__ void k_copyphi(const float* __restrict__ src, float* __restrict__ dst) {
    int idx = blockIdx.x * blockDim.x + threadIdx.x;
    if (idx < B_ * LS_) dst[idx] = src[idx];
}

// ---------------------------------------------------------------------------
// s = exp(i*phi); write s to ws, to output stack slot t, and (if write_x)
// refresh the s-columns of the MLP input.
// ---------------------------------------------------------------------------
__global__ void k_phase(const float* __restrict__ phi, float2* __restrict__ s_w,
                        _Float16* __restrict__ xh, float* __restrict__ out_stack,
                        int t, int write_x) {
    int idx = blockIdx.x * blockDim.x + threadIdx.x;   // 0..2047
    if (idx >= B_ * LS_) return;
    int b = idx / LS_, l = idx % LS_;
    float p = phi[idx];
    float sr = cosf(p), si = sinf(p);                  // MODULUS == 1.0
    s_w[idx] = make_float2(sr, si);
    size_t o = (((size_t)b * (NSTEP_ + 1) + t) * LS_ + l) * 2;
    out_stack[o] = sr; out_stack[o + 1] = si;
    if (write_x) {
        for (int m = 0; m < M_; ++m) {
            size_t row = (size_t)(b * M_ + m) * D_;
            xh[row + l]       = (_Float16)sr;
            xh[row + LS_ + l] = (_Float16)si;
        }
    }
}

// ---------------------------------------------------------------------------
// HBM-bound streamer: Gs[b,i,m] = sum_j G[b,i,j,m]*s[b,j]  (and same for H).
// Row G[b,i,:,:,:] is 64KB contiguous -> wave does float4 loads (512B/instr),
// lane p=lane&7 owns m-pair (2p,2p+1), lanes split j by (lane>>3),
// cross-lane sum via wave32 shfl_xor.
// ---------------------------------------------------------------------------
__global__ __launch_bounds__(256) void k_matvec(const float* __restrict__ G,
                                                const float* __restrict__ H,
                                                const float2* __restrict__ s_w,
                                                float* __restrict__ Gs,
                                                float* __restrict__ Hs) {
    int b    = blockIdx.y;
    int wave = threadIdx.x >> 5;
    int lane = threadIdx.x & 31;
    int i    = blockIdx.x * 8 + wave;

    __shared__ float2 s_sh[LS_];
    for (int k = threadIdx.x; k < LS_; k += 256) s_sh[k] = s_w[b * LS_ + k];
    __syncthreads();

    const float4* Gr = (const float4*)G + (size_t)(b * LS_ + i) * (LS_ * M_ * 2 / 4);
    const float4* Hr = (const float4*)H + (size_t)(b * LS_ + i) * (LS_ * M_ * 2 / 4);
    int p  = lane & 7;     // m-pair index: m0=2p, m1=2p+1
    int jo = lane >> 3;    // j sub-offset 0..3

    float g0r = 0.f, g0i = 0.f, g1r = 0.f, g1i = 0.f;
    float h0r = 0.f, h0i = 0.f, h1r = 0.f, h1i = 0.f;
    for (int j = 0; j < LS_; j += 4) {
        int jj = j + jo;
        float4 g  = Gr[jj * 8 + p];
        float4 h  = Hr[jj * 8 + p];
        float2 sv = s_sh[jj];
        g0r += g.x * sv.x - g.y * sv.y;  g0i += g.x * sv.y + g.y * sv.x;
        g1r += g.z * sv.x - g.w * sv.y;  g1i += g.z * sv.y + g.w * sv.x;
        h0r += h.x * sv.x - h.y * sv.y;  h0i += h.x * sv.y + h.y * sv.x;
        h1r += h.z * sv.x - h.w * sv.y;  h1i += h.z * sv.y + h.w * sv.x;
    }
    for (int off = 8; off < 32; off <<= 1) {
        g0r += __shfl_xor(g0r, off, 32); g0i += __shfl_xor(g0i, off, 32);
        g1r += __shfl_xor(g1r, off, 32); g1i += __shfl_xor(g1i, off, 32);
        h0r += __shfl_xor(h0r, off, 32); h0i += __shfl_xor(h0i, off, 32);
        h1r += __shfl_xor(h1r, off, 32); h1i += __shfl_xor(h1i, off, 32);
    }
    if (lane < 8) {
        size_t base = ((size_t)(b * LS_ + i) * M_ + 2 * lane) * 2;   // 16B aligned
        *(float4*)(Gs + base) = make_float4(g0r, g0i, g1r, g1i);
        *(float4*)(Hs + base) = make_float4(h0r, h0i, h1r, h1i);
    }
}

// ---------------------------------------------------------------------------
// sGs[b,m] = sum_i conj(s)*Gs ; sHs likewise. One wave per (b,m).
// ---------------------------------------------------------------------------
__global__ __launch_bounds__(32) void k_sdot(const float2* __restrict__ s_w,
                                             const float2* __restrict__ Gs2,
                                             const float2* __restrict__ Hs2,
                                             float2* __restrict__ sGs,
                                             float2* __restrict__ sHs) {
    int b = blockIdx.x >> 4, m = blockIdx.x & 15;
    int lane = threadIdx.x;
    float gr = 0.f, gi = 0.f, hr = 0.f, hi = 0.f;
    for (int i = lane; i < LS_; i += 32) {
        float2 sv = s_w[b * LS_ + i];
        float2 g  = Gs2[(size_t)(b * LS_ + i) * M_ + m];
        float2 h  = Hs2[(size_t)(b * LS_ + i) * M_ + m];
        gr += sv.x * g.x + sv.y * g.y;  gi += sv.x * g.y - sv.y * g.x;   // conj(s)*g
        hr += sv.x * h.x + sv.y * h.y;  hi += sv.x * h.y - sv.y * h.x;
    }
    for (int off = 16; off; off >>= 1) {
        gr += __shfl_xor(gr, off, 32); gi += __shfl_xor(gi, off, 32);
        hr += __shfl_xor(hr, off, 32); hi += __shfl_xor(hi, off, 32);
    }
    if (lane == 0) {
        sGs[b * M_ + m] = make_float2(gr, gi);
        sHs[b * M_ + m] = make_float2(hr, hi);
    }
}

// ---------------------------------------------------------------------------
// eta[b,i,m] = real(2/sGs^2) * imag((sHs*Gs - sGs*Hs)*conj(s))
// ---------------------------------------------------------------------------
__global__ void k_eta(const float2* __restrict__ s_w, const float2* __restrict__ Gs2,
                      const float2* __restrict__ Hs2, const float2* __restrict__ sGs,
                      const float2* __restrict__ sHs, float* __restrict__ eta) {
    int idx = blockIdx.x * blockDim.x + threadIdx.x;  // (b*LS + i)*M + m
    if (idx >= B_ * LS_ * M_) return;
    int m = idx & 15;
    int i = (idx >> 4) & (LS_ - 1);
    int b = idx >> 13;
    float2 sg = sGs[b * M_ + m], sh = sHs[b * M_ + m];
    float2 g  = Gs2[idx], h = Hs2[idx];
    float2 sv = s_w[b * LS_ + i];
    float tr = (sh.x * g.x - sh.y * g.y) - (sg.x * h.x - sg.y * h.y);
    float ti = (sh.x * g.y + sh.y * g.x) - (sg.x * h.y + sg.y * h.x);
    float inner = ti * sv.x - tr * sv.y;              // imag(t * conj(s))
    float cr = sg.x * sg.x - sg.y * sg.y;             // sGs^2
    float ci = 2.f * sg.x * sg.y;
    float coef = 2.f * cr / (cr * cr + ci * ci);      // real(2 / sGs^2)
    eta[idx] = coef * inner;
}

// ---------------------------------------------------------------------------
// MLP GEMM on the matrix pipe: h = relu(x[64,2560] * W1[t]^T[2560,128] + b1).
// One wave per 16x16 output tile; v_wmma_f32_16x16x32_f16 over 80 K-steps.
// A layout: lane row = lane&15; k chunks {k0+hiA .. +7} and {k0+hiA+16 .. +23},
//           hiA = (lane>=16)*8.  B layout: lane col = lane&15; same k chunks
//           as A (k0+hiB .. +7, k0+hiB+16 .. +23).  D: vgpr r -> m=r+(lane>=16)*8, n=lane&15.
// ---------------------------------------------------------------------------
__global__ __launch_bounds__(32) void k_mlp(const _Float16* __restrict__ xh,
                                            const _Float16* __restrict__ w1t,
                                            const float* __restrict__ bias1,
                                            float* __restrict__ hbuf) {
    int lane    = threadIdx.x;
    int rowtile = blockIdx.x & 3;      // 4 row tiles  (64 rows)
    int coltile = blockIdx.x >> 2;     // 8 col tiles  (128 cols)
    int arow = rowtile * 16 + (lane & 15);
    int bcol = coltile * 16 + (lane & 15);
    int hiA  = (lane >> 4) * 8;
    int hiB  = (lane >> 4) * 8;
    const _Float16* aptr = xh  + (size_t)arow * D_;
    const _Float16* bptr = w1t + (size_t)bcol * D_;

    v8f acc = {};
    for (int k0 = 0; k0 < D_; k0 += 32) {
        v8h a0 = *(const v8h*)(aptr + k0 + hiA);
        v8h a1 = *(const v8h*)(aptr + k0 + hiA + 16);
        v8h c0 = *(const v8h*)(bptr + k0 + hiB);
        v8h c1 = *(const v8h*)(bptr + k0 + hiB + 16);
        v16h A, Bv;
        for (int e = 0; e < 8; ++e) { A[e] = a0[e]; A[8 + e] = a1[e];
                                      Bv[e] = c0[e]; Bv[8 + e] = c1[e]; }
        acc = __builtin_amdgcn_wmma_f32_16x16x32_f16(
                  false, A, false, Bv, (short)0, acc, false, false);
    }
    int nn   = lane & 15;
    int mofs = (lane >> 4) * 8;
    for (int r = 0; r < 8; ++r) {
        int row = rowtile * 16 + r + mofs;
        int col = coltile * 16 + nn;
        float v = acc[r] + bias1[col];
        hbuf[(size_t)row * HID_ + col] = v > 0.f ? v : 0.f;
    }
}

// ---------------------------------------------------------------------------
// rho = sigmoid(h . W2 + b2); also emit rho_avg[b,t] = sum_m rho / M.
// ---------------------------------------------------------------------------
__global__ __launch_bounds__(64) void k_rho(const float* __restrict__ hbuf,
                                            const float* __restrict__ W2t,
                                            const float* __restrict__ b2, int t,
                                            float* __restrict__ rho,
                                            float* __restrict__ out_rho) {
    int r = threadIdx.x;               // 0..63
    float acc = b2[t];
    const float* hr = hbuf + (size_t)r * HID_;
    for (int k = 0; k < HID_; ++k) acc += hr[k] * W2t[k];
    float rv = 1.f / (1.f + expf(-acc));
    rho[r] = rv;
    __shared__ float sh[ROWS_];
    sh[r] = rv;
    __syncthreads();
    if (r < B_) {
        float s = 0.f;
        for (int m = 0; m < M_; ++m) s += sh[r * M_ + m];
        out_rho[r * NSTEP_ + t] = s / (float)M_;
    }
}

// ---------------------------------------------------------------------------
// phi[b,l] -= sum_m rho[b,m] * eta[b,l,m]
// ---------------------------------------------------------------------------
__global__ void k_phiupd(const float* __restrict__ rho, const float* __restrict__ eta,
                         float* __restrict__ phi) {
    int idx = blockIdx.x * blockDim.x + threadIdx.x;   // 0..2047
    if (idx >= B_ * LS_) return;
    int b = idx / LS_;
    const float* e  = eta + (size_t)idx * M_;
    const float* rb = rho + b * M_;
    float s = 0.f;
    for (int m = 0; m < M_; ++m) s += rb[m] * e[m];
    phi[idx] -= s;
}

// ---------------------------------------------------------------------------
extern "C" void kernel_launch(void* const* d_in, const int* in_sizes, int n_in,
                              void* d_out, int out_size, void* d_ws, size_t ws_size,
                              hipStream_t stream) {
    (void)in_sizes; (void)n_in; (void)out_size; (void)ws_size;
    const float* phi_in = (const float*)d_in[0];
    const float* w_in   = (const float*)d_in[1];
    const float* y_in   = (const float*)d_in[2];
    const float* G_in   = (const float*)d_in[3];
    const float* H_in   = (const float*)d_in[4];
    const float* W1_in  = (const float*)d_in[5];
    const float* b1_in  = (const float*)d_in[6];
    const float* W2_in  = (const float*)d_in[7];
    const float* b2_in  = (const float*)d_in[8];

    float* out     = (float*)d_out;                    // s_stack as (re,im) pairs
    float* out_rho = out + (size_t)B_ * (NSTEP_ + 1) * LS_ * 2;

    // ---- workspace carve-out (float units; 16B alignment holds throughout) ----
    float*  ws    = (float*)d_ws;
    float*  phi_w = ws;                                  // 2048
    float2* s_w   = (float2*)(ws + 2048);                // 2048 float2
    float*  Gs    = ws + 2048 + 4096;                    // 65536
    float*  Hs    = Gs + (size_t)B_ * LS_ * M_ * 2;      // 65536
    float2* sGs   = (float2*)(Hs + (size_t)B_ * LS_ * M_ * 2);  // 64 float2
    float2* sHs   = sGs + B_ * M_;                       // 64 float2
    float*  eta   = (float*)(sHs + B_ * M_);             // 32768
    float*  hbuf  = eta + (size_t)B_ * LS_ * M_;         // 8192
    float*  rho   = hbuf + (size_t)ROWS_ * HID_;         // 64
    _Float16* xh  = (_Float16*)(rho + 64);               // 64*2560 halfs
    _Float16* w1h = xh + (size_t)ROWS_ * D_;             // 10*128*2560 halfs

    // per-launch precompute (deterministic, graph-capture safe)
    k_w1half<<<(NSTEP_ * HID_ * D_ + 255) / 256, 256, 0, stream>>>(W1_in, w1h, NSTEP_ * HID_ * D_);
    k_xconst<<<(ROWS_ * 1536 + 255) / 256, 256, 0, stream>>>(w_in, y_in, xh);
    k_copyphi<<<(B_ * LS_ + 255) / 256, 256, 0, stream>>>(phi_in, phi_w);

    for (int t = 0; t < NSTEP_; ++t) {
        k_phase<<<(B_ * LS_ + 255) / 256, 256, 0, stream>>>(phi_w, s_w, xh, out, t, 1);
        k_matvec<<<dim3(LS_ / 8, B_), 256, 0, stream>>>(G_in, H_in, s_w, Gs, Hs);
        k_sdot<<<B_ * M_, 32, 0, stream>>>(s_w, (const float2*)Gs, (const float2*)Hs, sGs, sHs);
        k_eta<<<(B_ * LS_ * M_) / 256, 256, 0, stream>>>(s_w, (const float2*)Gs, (const float2*)Hs,
                                                         sGs, sHs, eta);
        k_mlp<<<32, 32, 0, stream>>>(xh, w1h + (size_t)t * HID_ * D_, b1_in + t * HID_, hbuf);
        k_rho<<<1, ROWS_, 0, stream>>>(hbuf, W2_in + t * HID_, b2_in, t, rho, out_rho);
        k_phiupd<<<(B_ * LS_ + 255) / 256, 256, 0, stream>>>(rho, eta, phi_w);
    }
    // final s into stack slot NSTEP
    k_phase<<<(B_ * LS_ + 255) / 256, 256, 0, stream>>>(phi_w, s_w, xh, out, NSTEP_, 0);
}
